// MultiheadAttention_26328149524482
// MI455X (gfx1250) — matrix operand order repaired and checked by reading, not compiled, not run on hardware
//
#include <hip/hip_runtime.h>
#include <hip/hip_bf16.h>

// ---------------------------------------------------------------------------
// Types for CDNA5 WMMA (wave32): V_WMMA_F32_16X16X32_BF16
// ---------------------------------------------------------------------------
typedef __bf16  bf16_t;
typedef __bf16  v16bf __attribute__((ext_vector_type(16)));
typedef __bf16  v8bf  __attribute__((ext_vector_type(8)));
typedef float   v8f   __attribute__((ext_vector_type(8)));

union FragU { v16bf f; v8bf h[2]; };

__device__ __forceinline__ v8f wmma_bf16(v16bf a, v16bf b, v8f c) {
    // (neg_a, A, neg_b, B, c_mod, C, reuse_a, reuse_b)
    return __builtin_amdgcn_wmma_f32_16x16x32_bf16(false, a, false, b,
                                                   (short)0, c, false, false);
}

// A-matrix 16x32 (16-bit): lane m = lane&15; per lane: K = [half*8 .. +7] and [half*8+16 .. +7]
__device__ __forceinline__ v16bf load_frag_a(const bf16_t* rowptr, int half) {
    FragU u;
    u.h[0] = *(const v8bf*)(rowptr + half * 8);
    u.h[1] = *(const v8bf*)(rowptr + half * 8 + 16);
    return u.f;
}
// B-matrix 32x16 (16-bit): lane n = lane&15; per lane: K = [half*8 .. +7] and [half*8+16 .. +7]
__device__ __forceinline__ v16bf load_frag_b(const bf16_t* rowptr, int half) {
    FragU u;
    u.h[0] = *(const v8bf*)(rowptr + half * 8);
    u.h[1] = *(const v8bf*)(rowptr + half * 8 + 16);
    return u.f;
}

// ---------------------------------------------------------------------------
// CDNA5 async Global -> LDS copy (tracked with ASYNCcnt).
//   global_load_async_to_lds_b128 vDstLds, v[gaddr:gaddr+1], off
// LDS destination address is the low 32 bits of the generic shared pointer
// ---------------------------------------------------------------------------
__device__ __forceinline__ void async_g2l_b128(void* lds_ptr, const void* gaddr) {
    uint32_t lds_addr = (uint32_t)(uintptr_t)lds_ptr;
    asm volatile("global_load_async_to_lds_b128 %0, %1, off"
                 :: "v"(lds_addr), "v"(gaddr)
                 : "memory");
}
__device__ __forceinline__ void wait_async0() {
    asm volatile("s_wait_asynccnt 0x0" ::: "memory");
}

// ---------------------------------------------------------------------------
// Problem constants
// ---------------------------------------------------------------------------
#define BATCH   2
#define SEQ     2048
#define DMODEL  1024
#define NHEAD   16
#define DK      64
#define MROWS   (BATCH * SEQ)          // 4096

// ---------------------------------------------------------------------------
// Kernel 1: fp32 -> bf16 conversion (grid-stride)
// ---------------------------------------------------------------------------
__global__ void cvt_f32_bf16(const float* __restrict__ in, bf16_t* __restrict__ out, int n) {
    int i = blockIdx.x * blockDim.x + threadIdx.x;
    int stride = gridDim.x * blockDim.x;
    for (; i < n; i += stride) out[i] = (bf16_t)in[i];
}

// ---------------------------------------------------------------------------
// Kernel 2/4: GEMM  out[m,n] = sum_k A[m,k] * W[n,k] + bias[n]
//   MODE 0: store bf16 head-split   [B,H,S,dk]   (Q, K)
//   MODE 1: store bf16 head-split-T [B,H,dk,S]   (V)
//   MODE 2: store fp32 row-major    [MROWS x D]  (output projection -> d_out)
// One wave computes a 32x32 tile (2x2 WMMA tiles). 4096 waves = 512 blocks.
// Weights (2 MB bf16) are L2-resident; direct fragment streaming is L2-bound.
// ---------------------------------------------------------------------------
template <int MODE>
__global__ __launch_bounds__(256)
void gemm_bf16_kernel(const bf16_t* __restrict__ A,
                      const bf16_t* __restrict__ W,
                      const float*  __restrict__ bias,
                      bf16_t* __restrict__ outb,
                      float*  __restrict__ outf) {
    constexpr int N = DMODEL, K = DMODEL;
    const int lane = threadIdx.x & 31;
    const int wid  = threadIdx.x >> 5;
    const int half = lane >> 4;
    const int l16  = lane & 15;

    const int gw   = blockIdx.x * 8 + wid;     // global wave id
    const int tn   = gw & 31;                  // N/32 = 32 tiles
    const int tm   = gw >> 5;
    const int row0 = tm * 32;
    const int col0 = tn * 32;

    v8f acc[2][2];
#pragma unroll
    for (int i = 0; i < 2; ++i)
#pragma unroll
        for (int j = 0; j < 2; ++j) acc[i][j] = (v8f)0.0f;

    for (int k0 = 0; k0 < K; k0 += 32) {
        v16bf afr[2], bfr[2];
#pragma unroll
        for (int t = 0; t < 2; ++t) {
            afr[t] = load_frag_a(A + (size_t)(row0 + t * 16 + l16) * K + k0, half);
            bfr[t] = load_frag_b(W + (size_t)(col0 + t * 16 + l16) * K + k0, half);
        }
#pragma unroll
        for (int i = 0; i < 2; ++i)
#pragma unroll
            for (int j = 0; j < 2; ++j)
                acc[i][j] = wmma_bf16(afr[i], bfr[j], acc[i][j]);
    }

    // Epilogue: C/D layout -> row = r + 8*half, col = l16 (within 16x16 tile)
#pragma unroll
    for (int j = 0; j < 2; ++j) {
        const int col = col0 + j * 16 + l16;
        const float bv = bias[col];
#pragma unroll
        for (int i = 0; i < 2; ++i) {
#pragma unroll
            for (int r = 0; r < 8; ++r) {
                const int row = row0 + i * 16 + r + 8 * half;
                const float v = acc[i][j][r] + bv;
                if (MODE == 2) {
                    outf[(size_t)row * N + col] = v;
                } else {
                    const int b = row >> 11, s = row & (SEQ - 1);
                    const int h = col >> 6,  d = col & (DK - 1);
                    size_t idx;
                    if (MODE == 0)
                        idx = (((size_t)b * NHEAD + h) * SEQ + s) * DK + d;
                    else
                        idx = (((size_t)b * NHEAD + h) * DK + d) * SEQ + s;
                    outb[idx] = (bf16_t)v;
                }
            }
        }
    }
}

// ---------------------------------------------------------------------------
// Kernel 3: flash attention (online softmax), bf16 WMMA, f32 accum.
//   Qh: [B,H,S,dk] bf16   Kh: [B,H,S,dk] bf16   Vt: [B,H,dk,S] bf16
//   Oc: [B,S,D]    bf16   (heads re-concatenated, feeds output projection)
// Block = 8 waves = 128 q rows of one (b,h). Per 64-key block, the workgroup
// cooperatively async-copies the K tile (8 KB, contiguous) and V^T tile
// (8 KB, 64 strided rows) into double-buffered LDS (global_load_async_to_lds,
// ASYNCcnt), overlapping the next copy with current-block WMMA + softmax.
// ---------------------------------------------------------------------------
__global__ __launch_bounds__(256)
void attn_kernel(const bf16_t* __restrict__ Qh,
                 const bf16_t* __restrict__ Kh,
                 const bf16_t* __restrict__ Vt,
                 bf16_t* __restrict__ Oc) {
    __shared__ bf16_t ldsK[2][64 * 64];    // [key n][dk]   8 KB per buffer
    __shared__ bf16_t ldsV[2][64 * 64];    // [d][local s]  8 KB per buffer
    __shared__ bf16_t lds_p[8][16 * 64];   // per-wave P tile (16 q x 64 keys)

    const int t    = threadIdx.x;
    const int lane = t & 31;
    const int wid  = t >> 5;
    const int half = lane >> 4;
    const int l16  = lane & 15;

    const int bh = blockIdx.x >> 4;        // 0..31  (b*16+h)
    const int sg = blockIdx.x & 15;        // strip group
    const int b  = bh >> 4;
    const int h  = bh & 15;
    const int q0 = (sg * 8 + wid) * 16;    // first q row of this wave

    const bf16_t* Qbase = Qh + ((size_t)bh * SEQ + q0) * DK;
    const bf16_t* Kbase = Kh + (size_t)bh * SEQ * DK;
    const bf16_t* Vbase = Vt + (size_t)bh * DK * SEQ;

    // Cooperative async prefetch of one 64-key block into LDS buffer `buf`.
    // K tile: rows j..j+63 of Kh[bh] are one contiguous 8 KB region.
    // V tile: 64 rows of V^T, 128 B each, row stride SEQ*2 B.
    auto prefetch = [&](int buf, int j) {
        {   // K: thread t copies 32 B at flat offset t*32
            const char* gk = (const char*)(Kbase + (size_t)j * DK) + t * 32;
            char*       lk = (char*)&ldsK[buf][0] + t * 32;
            async_g2l_b128(lk,      gk);
            async_g2l_b128(lk + 16, gk + 16);
        }
        {   // V: thread t copies 32 B of row (t>>2), chunk (t&3)
            const int row = t >> 2, ch = t & 3;
            const char* gv = (const char*)(Vbase + (size_t)row * SEQ + j) + ch * 32;
            char*       lv = (char*)&ldsV[buf][row * 64] + ch * 32;
            async_g2l_b128(lv,      gv);
            async_g2l_b128(lv + 16, gv + 16);
        }
    };

    // Q A-fragments for K-steps k0 = 0, 32 (dk = 64): loaded once from global.
    v16bf qa[2];
#pragma unroll
    for (int kk = 0; kk < 2; ++kk)
        qa[kk] = load_frag_a(Qbase + (size_t)l16 * DK + kk * 32, half);

    v8f o[4];
#pragma unroll
    for (int c = 0; c < 4; ++c) o[c] = (v8f)0.0f;
    float mr[8], lr[8];
#pragma unroll
    for (int r = 0; r < 8; ++r) { mr[r] = -1e30f; lr[r] = 0.0f; }

    const float scale = 0.125f;            // 1/sqrt(64)

    prefetch(0, 0);
    int cur = 0;

    for (int j = 0; j < SEQ; j += 64) {    // key blocks of 64
        wait_async0();                     // buf[cur] resident (per-wave cnt)
        __syncthreads();                   // all waves' copies visible; prior
                                           // reads of buf[cur^1] finished
        if (j + 64 < SEQ) prefetch(cur ^ 1, j + 64);   // overlap next copy

        // ---- scores: S16x64 = Q(16x64) @ K_block^T (K from LDS) --------
        v8f sacc[4];
#pragma unroll
        for (int c = 0; c < 4; ++c) sacc[c] = (v8f)0.0f;
#pragma unroll
        for (int kk = 0; kk < 2; ++kk) {
#pragma unroll
            for (int c = 0; c < 4; ++c) {
                v16bf kb = load_frag_b(&ldsK[cur][(c * 16 + l16) * 64 + kk * 32], half);
                sacc[c] = wmma_bf16(qa[kk], kb, sacc[c]);
            }
        }

        // ---- online softmax over the 64 new columns -------------------
#pragma unroll
        for (int r = 0; r < 8; ++r) {
            float bm = -1e30f;
#pragma unroll
            for (int c = 0; c < 4; ++c) {
                sacc[c][r] *= scale;
                bm = fmaxf(bm, sacc[c][r]);
            }
            // reduce over the 16 lanes holding this row (stays in half)
#pragma unroll
            for (int msk = 1; msk < 16; msk <<= 1)
                bm = fmaxf(bm, __shfl_xor(bm, msk, 32));
            const float mn    = fmaxf(mr[r], bm);
            const float alpha = __expf(mr[r] - mn);
            mr[r] = mn;
            lr[r] *= alpha;
#pragma unroll
            for (int c = 0; c < 4; ++c) o[c][r] *= alpha;

            float rs = 0.0f;
#pragma unroll
            for (int c = 0; c < 4; ++c) {
                const float pv = __expf(sacc[c][r] - mn);
                rs += pv;
                lds_p[wid][(r + 8 * half) * 64 + c * 16 + l16] = (bf16_t)pv;
            }
#pragma unroll
            for (int msk = 1; msk < 16; msk <<= 1)
                rs += __shfl_xor(rs, msk, 32);
            lr[r] += rs;
        }

        // ---- O += P(16x64) @ V_block(64 x dk)  (P, V from LDS) --------
#pragma unroll
        for (int kk = 0; kk < 2; ++kk) {
            v16bf pa = load_frag_a(&lds_p[wid][(size_t)l16 * 64 + kk * 32], half);
#pragma unroll
            for (int c = 0; c < 4; ++c) {
                v16bf vb = load_frag_b(&ldsV[cur][(c * 16 + l16) * 64 + kk * 32], half);
                o[c] = wmma_bf16(pa, vb, o[c]);
            }
        }

        cur ^= 1;
    }

    // ---- epilogue: normalize, write concat layout [B,S,D] -------------
#pragma unroll
    for (int r = 0; r < 8; ++r) {
        const float inv = 1.0f / lr[r];
        const int s_row = q0 + r + 8 * half;
        const size_t base = ((size_t)b * SEQ + s_row) * DMODEL + h * DK;
#pragma unroll
        for (int c = 0; c < 4; ++c)
            Oc[base + c * 16 + l16] = (bf16_t)(o[c][r] * inv);
    }
}

// ---------------------------------------------------------------------------
// Host side
// ---------------------------------------------------------------------------
extern "C" void kernel_launch(void* const* d_in, const int* in_sizes, int n_in,
                              void* d_out, int out_size, void* d_ws, size_t ws_size,
                              hipStream_t stream) {
    const float* q   = (const float*)d_in[0];
    const float* k   = (const float*)d_in[1];
    const float* v   = (const float*)d_in[2];
    const float* W_q = (const float*)d_in[3];
    const float* b_q = (const float*)d_in[4];
    const float* W_k = (const float*)d_in[5];
    const float* b_k = (const float*)d_in[6];
    const float* W_v = (const float*)d_in[7];
    const float* b_v = (const float*)d_in[8];
    const float* W_o = (const float*)d_in[9];
    const float* b_o = (const float*)d_in[10];

    char* ws = (char*)d_ws;
    const size_t MB = 1ull << 20;
    bf16_t* WQ = (bf16_t*)(ws + 0 * MB);   // 2 MB each (1024x1024 bf16)
    bf16_t* WK = (bf16_t*)(ws + 2 * MB);
    bf16_t* WV = (bf16_t*)(ws + 4 * MB);
    bf16_t* WO = (bf16_t*)(ws + 6 * MB);
    bf16_t* Xq = (bf16_t*)(ws + 8 * MB);   // 8 MB each (4096x1024 bf16)
    bf16_t* Xk = (bf16_t*)(ws + 16 * MB);
    bf16_t* Xv = (bf16_t*)(ws + 24 * MB);
    bf16_t* Qh = (bf16_t*)(ws + 32 * MB);  // [B,H,S,dk]
    bf16_t* Kh = (bf16_t*)(ws + 40 * MB);  // [B,H,S,dk]
    bf16_t* Vt = (bf16_t*)(ws + 48 * MB);  // [B,H,dk,S]
    bf16_t* Oc = (bf16_t*)(ws + 56 * MB);  // [B,S,D]

    const int nAct = MROWS * DMODEL;       // 4 Mi elements
    const int nW   = DMODEL * DMODEL;      // 1 Mi elements
    cvt_f32_bf16<<<1024, 256, 0, stream>>>(q,   Xq, nAct);
    cvt_f32_bf16<<<1024, 256, 0, stream>>>(k,   Xk, nAct);
    cvt_f32_bf16<<<1024, 256, 0, stream>>>(v,   Xv, nAct);
    cvt_f32_bf16<<<512,  256, 0, stream>>>(W_q, WQ, nW);
    cvt_f32_bf16<<<512,  256, 0, stream>>>(W_k, WK, nW);
    cvt_f32_bf16<<<512,  256, 0, stream>>>(W_v, WV, nW);
    cvt_f32_bf16<<<512,  256, 0, stream>>>(W_o, WO, nW);

    // Projections: 4096 waves -> 512 blocks of 8 waves
    gemm_bf16_kernel<0><<<512, 256, 0, stream>>>(Xq, WQ, b_q, Qh, nullptr);
    gemm_bf16_kernel<0><<<512, 256, 0, stream>>>(Xk, WK, b_k, Kh, nullptr);
    gemm_bf16_kernel<1><<<512, 256, 0, stream>>>(Xv, WV, b_v, Vt, nullptr);

    // Flash attention: 32 (b,h) x 16 strip-groups = 512 blocks
    attn_kernel<<<512, 256, 0, stream>>>(Qh, Kh, Vt, Oc);

    // Output projection -> fp32 d_out
    gemm_bf16_kernel<2><<<512, 256, 0, stream>>>(Oc, WO, b_o, nullptr, (float*)d_out);
}
